// ShiftWindowAttentionBlock_3324304687556
// MI455X (gfx1250) — compile-verified, matrix operand order checked by reading, not run on hardware
//
#include <hip/hip_runtime.h>

// ---------------------------------------------------------------------------
// ShiftWindowAttentionBlock for MI455X (gfx1250): f16 WMMA GEMM pipeline
//   all matmuls -> v_wmma_f32_16x16x32_f16 (f32 accumulate)
// ---------------------------------------------------------------------------

typedef __attribute__((ext_vector_type(16))) _Float16 v16h;
typedef __attribute__((ext_vector_type(8)))  _Float16 v8h;
typedef __attribute__((ext_vector_type(8)))  float    v8f;

#define LOG_MAX 4.6051701859880913680f   // ln(1/0.01)

// ---------------------------------------------------------------------------
// fp32 -> fp16 conversion (grid-stride)
// ---------------------------------------------------------------------------
__global__ void f32_to_f16_kernel(const float* __restrict__ src,
                                  _Float16* __restrict__ dst, long n) {
  long i = (long)blockIdx.x * blockDim.x + threadIdx.x;
  long stride = (long)gridDim.x * blockDim.x;
  for (; i < n; i += stride) dst[i] = (_Float16)src[i];
}

// ---------------------------------------------------------------------------
// Tiled WMMA GEMM:  C(M,N) = op(A) * op(B)  (+ epilogue), f16 in / f32 acc
//   TA==0: A element (m,k) at A[m*lda + k]      (row-major M x K)
//   TA==1: A element (m,k) at A[k*lda + m]      (memory K x M)
//   TB==0: B element (k,n) at B[n*ldb + k]      (memory N x K, i.e. A*B^T)
//   TB==1: B element (k,n) at B[k*ldb + n]      (memory K x N)
//   EPI 1: +bias[n], relu, store f16    (GEMM1)
//   EPI 2: +bias[n],        store f16   (GEMM2)
//   EPI 3: *exp(min(lscale[n],LOG_MAX)) store f32  (GEMM3)
//   EPI 4: plain            store f32   (GEMM4)
// Block tile 128x128x32, 256 threads = 8 waves (4x2), wave tile 32x64.
// ---------------------------------------------------------------------------
template <int TA, int TB, int EPI>
__global__ void __launch_bounds__(256)
gemm_f16_wmma(const _Float16* __restrict__ A, const _Float16* __restrict__ Bm,
              float* __restrict__ Cf, _Float16* __restrict__ Ch,
              const float* __restrict__ bias, const float* __restrict__ lscale,
              int M, int N, int K, int lda, int ldb, int ldc,
              long batchA, long batchB, long batchC) {
  constexpr int BM = 128, BN = 128, BK = 32;
  constexpr int PITCH = 40;  // halfs per LDS row: 80B = 20 dwords, conflict-free
  __shared__ _Float16 sA[BM * PITCH];
  __shared__ _Float16 sB[BN * PITCH];

  const int tid  = threadIdx.x;
  const int lane = tid & 31;
  const int wave = tid >> 5;
  const int wm   = wave & 3;   // 4 waves along M (32 rows each)
  const int wn   = wave >> 2;  // 2 waves along N (64 cols each)
  const int hl   = lane >> 4;  // lane half (K-chunk select)
  const int l16  = lane & 15;

  const int  bm = blockIdx.x * BM;
  const int  bn = blockIdx.y * BN;
  const long zb = blockIdx.z;
  A  += zb * batchA;
  Bm += zb * batchB;
  if constexpr (EPI == 1 || EPI == 2) Ch += zb * batchC;
  else                                Cf += zb * batchC;

  v8f acc[2][4] = {};

  const int kblocks = K / BK;
  for (int kb = 0; kb < kblocks; ++kb) {
    const int k0 = kb * BK;

    // ---- stage A tile into sA[m][k] ----
    if constexpr (TA == 0) {
      const int row = tid >> 1;
      const int ko  = (tid & 1) * 16;
      const _Float16* src = A + (long)(bm + row) * lda + (k0 + ko);
      v8h* d = (v8h*)&sA[row * PITCH + ko];
      d[0] = ((const v8h*)src)[0];
      d[1] = ((const v8h*)src)[1];
      if (kb + 1 < kblocks) __builtin_prefetch(src + BK, 0, 0);
    } else {
      const int kr = tid >> 3;          // 0..31
      const int mo = (tid & 7) * 16;    // 0..112
      const _Float16* src = A + (long)(k0 + kr) * lda + (bm + mo);
#pragma unroll
      for (int i = 0; i < 16; ++i) sA[(mo + i) * PITCH + kr] = src[i];
    }
    // ---- stage B tile into sB[n][k] ----
    if constexpr (TB == 0) {
      const int row = tid >> 1;
      const int ko  = (tid & 1) * 16;
      const _Float16* src = Bm + (long)(bn + row) * ldb + (k0 + ko);
      v8h* d = (v8h*)&sB[row * PITCH + ko];
      d[0] = ((const v8h*)src)[0];
      d[1] = ((const v8h*)src)[1];
      if (kb + 1 < kblocks) __builtin_prefetch(src + BK, 0, 0);
    } else {
      const int kr = tid >> 3;
      const int no = (tid & 7) * 16;
      const _Float16* src = Bm + (long)(k0 + kr) * ldb + (bn + no);
#pragma unroll
      for (int i = 0; i < 16; ++i) sB[(no + i) * PITCH + kr] = src[i];
    }
    __syncthreads();

    // ---- fragments per documented 16-bit 16x32 layout ----
    // lane half 0: v0..3 <- k0..7,  v4..7 <- k16..23
    // lane half 1: v0..3 <- k8..15, v4..7 <- k24..31
    v16h afrag[2], bfrag[4];
#pragma unroll
    for (int mi = 0; mi < 2; ++mi) {
      const _Float16* p = &sA[(wm * 32 + mi * 16 + l16) * PITCH + hl * 8];
      v8h lo = *(const v8h*)p;
      v8h hi = *(const v8h*)(p + 16);
      afrag[mi] = __builtin_shufflevector(lo, hi, 0, 1, 2, 3, 4, 5, 6, 7, 8, 9,
                                          10, 11, 12, 13, 14, 15);
    }
#pragma unroll
    for (int ni = 0; ni < 4; ++ni) {
      const _Float16* p = &sB[(wn * 64 + ni * 16 + l16) * PITCH + hl * 8];
      v8h lo = *(const v8h*)p;
      v8h hi = *(const v8h*)(p + 16);
      bfrag[ni] = __builtin_shufflevector(lo, hi, 0, 1, 2, 3, 4, 5, 6, 7, 8, 9,
                                          10, 11, 12, 13, 14, 15);
    }
#pragma unroll
    for (int mi = 0; mi < 2; ++mi)
#pragma unroll
      for (int ni = 0; ni < 4; ++ni)
        acc[mi][ni] = __builtin_amdgcn_wmma_f32_16x16x32_f16(
            false, afrag[mi], false, bfrag[ni], (short)0, acc[mi][ni],
            false, false);
    __syncthreads();
  }

  // ---- epilogue: D layout lane=(n=l16), vgpr v -> m = v + 8*half ----
#pragma unroll
  for (int ni = 0; ni < 4; ++ni) {
    const int gcol = bn + wn * 64 + ni * 16 + l16;
    float badd = 0.0f, smul = 1.0f;
    if constexpr (EPI == 1 || EPI == 2) badd = bias[gcol];
    if constexpr (EPI == 3) smul = __expf(fminf(lscale[gcol], LOG_MAX));
#pragma unroll
    for (int mi = 0; mi < 2; ++mi) {
      const int rbase = bm + wm * 32 + mi * 16 + hl * 8;
#pragma unroll
      for (int v = 0; v < 8; ++v) {
        const long idx = (long)(rbase + v) * ldc + gcol;
        float val = acc[mi][ni][v];
        if constexpr (EPI == 1) {
          val = fmaxf(val + badd, 0.0f);
          Ch[idx] = (_Float16)val;
        } else if constexpr (EPI == 2) {
          Ch[idx] = (_Float16)(val + badd);
        } else if constexpr (EPI == 3) {
          Cf[idx] = val * smul;
        } else {
          Cf[idx] = val;
        }
      }
    }
  }
}

// ---------------------------------------------------------------------------
// softmax over dim=1 of temp (B,N,N): per (b,m) column; writes f16 probs
// ---------------------------------------------------------------------------
__global__ void softmax_dim1_kernel(const float* __restrict__ T,
                                    _Float16* __restrict__ P, int Nn) {
  const int m = blockIdx.x * blockDim.x + threadIdx.x;
  const long base = (long)blockIdx.y * Nn * Nn + m;
  float mx = -3.402823466e38f;
  for (int n = 0; n < Nn; ++n) mx = fmaxf(mx, T[base + (long)n * Nn]);
  float s = 0.0f;
  for (int n = 0; n < Nn; ++n) s += __expf(T[base + (long)n * Nn] - mx);
  const float inv = 1.0f / s;
  for (int n = 0; n < Nn; ++n)
    P[base + (long)n * Nn] = (_Float16)(__expf(T[base + (long)n * Nn] - mx) * inv);
}

// ---------------------------------------------------------------------------
// LayerNorm over last dim, in-place on (B*C, N) f32 rows
// ---------------------------------------------------------------------------
__global__ void layernorm_rows_kernel(float* __restrict__ out,
                                      const float* __restrict__ w,
                                      const float* __restrict__ b, int Nn) {
  __shared__ float s1[256], s2[256];
  float* p = out + (long)blockIdx.x * Nn;
  const int tid = threadIdx.x;
  float sum = 0.0f, sq = 0.0f;
  for (int i = tid; i < Nn; i += 256) {
    const float v = p[i];
    sum += v;
    sq += v * v;
  }
  s1[tid] = sum;
  s2[tid] = sq;
  __syncthreads();
  for (int off = 128; off > 0; off >>= 1) {
    if (tid < off) {
      s1[tid] += s1[tid + off];
      s2[tid] += s2[tid + off];
    }
    __syncthreads();
  }
  const float mu  = s1[0] / (float)Nn;
  const float var = s2[0] / (float)Nn - mu * mu;
  const float inv = rsqrtf(var + 1e-5f);
  for (int i = tid; i < Nn; i += 256) {
    p[i] = (p[i] - mu) * inv * w[i] + b[i];
  }
}

// ---------------------------------------------------------------------------
extern "C" void kernel_launch(void* const* d_in, const int* in_sizes, int n_in,
                              void* d_out, int out_size, void* d_ws,
                              size_t ws_size, hipStream_t stream) {
  (void)in_sizes; (void)n_in; (void)out_size; (void)ws_size;

  const int NT = 4096;        // spatial tokens N
  const int F2 = 16384;       // mlp input dim
  const int Bb = 8, Cc = 512;
  const long MBC = (long)Bb * Cc;  // 4096 folded batch*channel rows

  const float* x           = (const float*)d_in[0];
  const float* feature     = (const float*)d_in[1];
  const float* fc1_w       = (const float*)d_in[2];
  const float* fc1_b       = (const float*)d_in[3];
  const float* fc2_w       = (const float*)d_in[4];
  const float* fc2_b       = (const float*)d_in[5];
  const float* logit_scale = (const float*)d_in[6];
  const float* norm_w      = (const float*)d_in[7];
  const float* norm_b      = (const float*)d_in[8];
  float* out = (float*)d_out;

  // ---- workspace layout (256B aligned) ----
  char* wsp = (char*)d_ws;
  auto alloc = [&](size_t bytes) -> char* {
    char* p = wsp;
    wsp += (bytes + 255) & ~(size_t)255;
    return p;
  };
  _Float16* feat_h = (_Float16*)alloc(MBC * F2 * sizeof(_Float16));
  _Float16* w1_h   = (_Float16*)alloc((long)NT * F2 * sizeof(_Float16));
  _Float16* w2_h   = (_Float16*)alloc((long)NT * NT * sizeof(_Float16));
  _Float16* x_h    = (_Float16*)alloc(MBC * NT * sizeof(_Float16));
  _Float16* h_h    = (_Float16*)alloc(MBC * NT * sizeof(_Float16));
  _Float16* f_h    = (_Float16*)alloc(MBC * NT * sizeof(_Float16));
  float*    temp_f = (float*)alloc((long)Bb * NT * NT * sizeof(float));
  _Float16* P_h    = (_Float16*)alloc((long)Bb * NT * NT * sizeof(_Float16));

  // ---- convert inputs to f16 once ----
  f32_to_f16_kernel<<<4096, 256, 0, stream>>>(feature, feat_h, MBC * F2);
  f32_to_f16_kernel<<<4096, 256, 0, stream>>>(fc1_w, w1_h, (long)NT * F2);
  f32_to_f16_kernel<<<2048, 256, 0, stream>>>(fc2_w, w2_h, (long)NT * NT);
  f32_to_f16_kernel<<<2048, 256, 0, stream>>>(x, x_h, MBC * NT);

  // GEMM1: h = relu(feature(4096x16384) * fc1_w^T(16384x4096) + b1) -> f16
  gemm_f16_wmma<0, 0, 1><<<dim3(32, 32, 1), 256, 0, stream>>>(
      feat_h, w1_h, nullptr, h_h, fc1_b, nullptr,
      (int)MBC, NT, F2, F2, F2, NT, 0, 0, 0);

  // GEMM2: feat = h(4096x4096) * fc2_w^T(4096x4096) + b2 -> f16
  gemm_f16_wmma<0, 0, 2><<<dim3(32, 32, 1), 256, 0, stream>>>(
      h_h, w2_h, nullptr, f_h, fc2_b, nullptr,
      (int)MBC, NT, NT, NT, NT, NT, 0, 0, 0);

  // GEMM3 (batched over b): temp[b](n,m) = sum_c feat[b](c,n)*x[b](c,m),
  //                         * exp(min(logit_scale[m], LOG_MAX)) -> f32
  gemm_f16_wmma<1, 1, 3><<<dim3(32, 32, Bb), 256, 0, stream>>>(
      f_h, x_h, temp_f, nullptr, nullptr, logit_scale,
      NT, NT, Cc, NT, NT, NT,
      (long)Cc * NT, (long)Cc * NT, (long)NT * NT);

  // softmax over dim=1 (columns of temp[b]) -> f16 probabilities
  softmax_dim1_kernel<<<dim3(NT / 256, Bb), 256, 0, stream>>>(temp_f, P_h, NT);

  // GEMM4 (batched): out[b](c,m) = sum_n x[b](c,n) * P[b](n,m) -> f32 (d_out)
  gemm_f16_wmma<0, 1, 4><<<dim3(4, 32, Bb), 256, 0, stream>>>(
      x_h, P_h, out, nullptr, nullptr, nullptr,
      Cc, NT, NT, NT, NT, NT,
      (long)Cc * NT, (long)NT * NT, (long)Cc * NT);

  // LayerNorm over last dim, in-place on d_out rows
  layernorm_rows_kernel<<<(int)MBC, 256, 0, stream>>>(out, norm_w, norm_b, NT);
}
